// SpatialScaffold_30253749633090
// MI455X (gfx1250) — compile-verified, matrix operand order checked by reading, not run on hardware
//
#include <hip/hip_runtime.h>

// ---- CDNA5 WMMA types ----
typedef _Float16 v16h __attribute__((ext_vector_type(16)));
typedef _Float16 h8   __attribute__((ext_vector_type(8)));
typedef float    v8f  __attribute__((ext_vector_type(8)));
typedef float    v4f  __attribute__((ext_vector_type(4)));

#define LAT 128   // LATENT == HIDDEN == 128

// Build a 16-bit A/B fragment (v16h, 8 VGPRs) from f16 LDS where the needed
// K-run is contiguous:  elements 0..7  = p[0..7]   (K = k0 + hl*8 + i)
//                       elements 8..15 = p[16..23] (K = k0 + 16 + hl*8 + i)
__device__ __forceinline__ v16h frag_from_lds(const _Float16* p) {
  h8 lo = *(const h8*)(p);
  h8 hi = *(const h8*)(p + 16);
  v16h r;
#pragma unroll
  for (int i = 0; i < 8; ++i) { r[i] = lo[i]; r[i + 8] = hi[i]; }
  return r;
}

// Same fragment built from a global f32 row (b128 loads), with optional scale,
// converted to f16 in-register (v_cvt_pk_f16_f32 pairs).
__device__ __forceinline__ v16h frag_from_global_f32(const float* p, float scale) {
  v4f a0 = *(const v4f*)(p);
  v4f a1 = *(const v4f*)(p + 4);
  v4f b0 = *(const v4f*)(p + 16);
  v4f b1 = *(const v4f*)(p + 20);
  v16h r;
#pragma unroll
  for (int i = 0; i < 4; ++i) {
    r[i]      = (_Float16)(a0[i] * scale);
    r[i + 4]  = (_Float16)(a1[i] * scale);
    r[i + 8]  = (_Float16)(b0[i] * scale);
    r[i + 12] = (_Float16)(b1[i] * scale);
  }
  return r;
}

// ---------------------------------------------------------------------------
// Kernel 0: zero agg + degree scratch (must re-zero every launch).
__global__ void SpatialScaffold_zero_kernel(float* __restrict__ p, size_t n) {
  size_t i = (size_t)blockIdx.x * blockDim.x + threadIdx.x;
  size_t stride = (size_t)gridDim.x * blockDim.x;
  for (; i < n; i += stride) p[i] = 0.0f;
}

// ---------------------------------------------------------------------------
// Kernel 1: scatter-add. One wave per edge; each lane handles 4 features
// (b128 gather of u_st[col], 4 global f32 atomics into agg[row]). agg fits in
// the 192MB L2, so the atomics resolve there.
__global__ __launch_bounds__(256) void SpatialScaffold_scatter_kernel(
    const float* __restrict__ u, const int* __restrict__ row,
    const int* __restrict__ col, float* __restrict__ agg,
    float* __restrict__ deg, int nEdges) {
  int gid  = blockIdx.x * 256 + threadIdx.x;
  int e    = gid >> 5;
  int lane = gid & 31;
  if (e >= nEdges) return;
  int r = row[e];
  int c = col[e];
  v4f vals = *(const v4f*)(u + (size_t)c * LAT + lane * 4);
  float* dst = agg + (size_t)r * LAT + lane * 4;
#pragma unroll
  for (int i = 0; i < 4; ++i) atomicAdd(dst + i, vals[i]);
  if (lane == 0) atomicAdd(deg + r, 1.0f);
}

// ---------------------------------------------------------------------------
// Kernel 2 (fused everything):
//   h      = leaky_relu(u@W1 + b1, 0.2)          (staged per-wave in LDS, f16)
//   fused  = h@W2 + b2                           (held in registers)
//   sb     = leaky_relu((agg/deg)@Wp + bp, 0.2)  (held in registers)
//   out    = fused + sb                          (single b32 store per element)
// 8 waves/block, each wave owns a 16-row tile. Writing out exactly once saves
// ~200MB of HBM RMW traffic vs a split fused/spatial pipeline.
__global__ __launch_bounds__(256) void SpatialScaffold_fused_kernel(
    const float* __restrict__ u, const float* __restrict__ W1,
    const float* __restrict__ b1, const float* __restrict__ W2,
    const float* __restrict__ b2, const float* __restrict__ Wp,
    const float* __restrict__ bp, const float* __restrict__ agg,
    const float* __restrict__ deg, float* __restrict__ out, int nTiles) {
  __shared__ _Float16 sW1T[LAT * LAT];   // W1^T as f16: [j][k] contiguous in k
  __shared__ _Float16 sW2T[LAT * LAT];   // W2^T as f16
  __shared__ _Float16 sWpT[LAT * LAT];   // Wp^T as f16
  __shared__ _Float16 sH[8 * 16 * LAT];  // per-wave h tile (16 x 128, f16)

  // Cooperative transpose of the weights into LDS (once per block).
  for (int idx = threadIdx.x; idx < LAT * LAT; idx += 256) {
    int k = idx >> 7, j = idx & 127;
    sW1T[j * LAT + k] = (_Float16)W1[idx];
    sW2T[j * LAT + k] = (_Float16)W2[idx];
    sWpT[j * LAT + k] = (_Float16)Wp[idx];
  }
  __syncthreads();

  const int wave = threadIdx.x >> 5;
  const int lane = threadIdx.x & 31;
  const int m  = lane & 15;   // A-frag row / B-frag column / C-frag column
  const int hl = lane >> 4;   // half-wave selector
  _Float16* hrow = &sH[wave * (16 * LAT)];

  for (int t = blockIdx.x * 8 + wave; t < nTiles; t += gridDim.x * 8) {
    const int n0 = t * 16;

    // A fragments of u and of the normalized aggregate (4 K-chunks of 32).
    float d = deg[n0 + m];
    float invd = 1.0f / fmaxf(d, 1e-8f);
    v16h afr[4], gfr[4];
#pragma unroll
    for (int kc = 0; kc < 4; ++kc) {
      afr[kc] = frag_from_global_f32(u   + (size_t)(n0 + m) * LAT + kc * 32 + hl * 8, 1.0f);
      gfr[kc] = frag_from_global_f32(agg + (size_t)(n0 + m) * LAT + kc * 32 + hl * 8, invd);
    }

    // GEMM1: h = leaky_relu(u@W1 + b1)  -> staged to LDS as f16.
#pragma unroll
    for (int jb = 0; jb < 8; ++jb) {
      v8f c = {};
#pragma unroll
      for (int kc = 0; kc < 4; ++kc) {
        v16h b = frag_from_lds(&sW1T[(jb * 16 + m) * LAT + kc * 32 + hl * 8]);
        c = __builtin_amdgcn_wmma_f32_16x16x32_f16(false, afr[kc], false, b,
                                                   (short)0, c, false, false);
      }
      const float bias = b1[jb * 16 + m];
#pragma unroll
      for (int v = 0; v < 8; ++v) {
        float x = c[v] + bias;
        x = x > 0.0f ? x : 0.2f * x;                       // leaky_relu(0.2)
        hrow[(v + 8 * hl) * LAT + jb * 16 + m] = (_Float16)x;
      }
    }
    // Same-wave LDS ops are processed in order: RAW through sH is safe.

    // A fragments of h from LDS (contiguous f16 -> ds_load_b128).
    v16h hfr[4];
#pragma unroll
    for (int kc = 0; kc < 4; ++kc)
      hfr[kc] = frag_from_lds(&hrow[m * LAT + kc * 32 + hl * 8]);

    // GEMM2 + GEMM3 per output j-block, combine, single store.
#pragma unroll
    for (int jb = 0; jb < 8; ++jb) {
      v8f c2 = {};   // fused   = h @ W2
      v8f c3 = {};   // spatial = (agg/deg) @ Wp
#pragma unroll
      for (int kc = 0; kc < 4; ++kc) {
        v16h b2f = frag_from_lds(&sW2T[(jb * 16 + m) * LAT + kc * 32 + hl * 8]);
        c2 = __builtin_amdgcn_wmma_f32_16x16x32_f16(false, hfr[kc], false, b2f,
                                                    (short)0, c2, false, false);
        v16h bpf = frag_from_lds(&sWpT[(jb * 16 + m) * LAT + kc * 32 + hl * 8]);
        c3 = __builtin_amdgcn_wmma_f32_16x16x32_f16(false, gfr[kc], false, bpf,
                                                    (short)0, c3, false, false);
      }
      const float bias2 = b2[jb * 16 + m];
      const float biasp = bp[jb * 16 + m];
#pragma unroll
      for (int v = 0; v < 8; ++v) {
        float sb = c3[v] + biasp;
        sb = sb > 0.0f ? sb : 0.2f * sb;                   // leaky_relu(0.2)
        out[(size_t)(n0 + v + 8 * hl) * LAT + jb * 16 + m] = c2[v] + bias2 + sb;
      }
    }
  }
}

// ---------------------------------------------------------------------------
extern "C" void kernel_launch(void* const* d_in, const int* in_sizes, int n_in,
                              void* d_out, int out_size, void* d_ws, size_t ws_size,
                              hipStream_t stream) {
  const float* u  = (const float*)d_in[0];   // [N, 128]
  const float* W1 = (const float*)d_in[1];   // [128, 128]
  const float* b1 = (const float*)d_in[2];   // [128]
  const float* W2 = (const float*)d_in[3];   // [128, 128]
  const float* b2 = (const float*)d_in[4];   // [128]
  const float* Wp = (const float*)d_in[5];   // [128, 128]
  const float* bp = (const float*)d_in[6];   // [128]
  const int*   row = (const int*)d_in[7];    // [E]
  const int*   col = (const int*)d_in[8];    // [E]

  const int N  = in_sizes[0] / LAT;          // 200000
  const int nE = in_sizes[7];                // 1.2M
  const int nTiles = N / 16;                 // 12500 (N divisible by 16)

  float* agg = (float*)d_ws;                 // [N, 128]
  float* deg = agg + (size_t)N * LAT;        // [N]
  float* out = (float*)d_out;

  // 0) zero scratch (agg + degree) every call
  SpatialScaffold_zero_kernel<<<2048, 256, 0, stream>>>(agg, (size_t)N * (LAT + 1));
  // 1) scatter-add: agg[row] += u[col], deg[row] += 1
  SpatialScaffold_scatter_kernel<<<(nE * 32 + 255) / 256, 256, 0, stream>>>(
      u, row, col, agg, deg, nE);
  // 2) fully fused MLP + spatial bias -> out (written exactly once)
  SpatialScaffold_fused_kernel<<<512, 256, 0, stream>>>(
      u, W1, b1, W2, b2, Wp, bp, agg, deg, out, nTiles);
}
